// Grok1MoE_19705309954124
// MI455X (gfx1250) — compile-verified, matrix operand order checked by reading, not run on hardware
//
#include <hip/hip_runtime.h>
#include <hip/hip_bf16.h>

// ---------------------------------------------------------------------------
// Grok1 MoE (B=4,S=1024,H=1024,E=8,I=2048,top-2) for MI455X (gfx1250, wave32)
// bf16 WMMA (v_wmma_f32_16x16x32_bf16) expert GEMMs, fp32 accumulate.
// bf16 weights (96MB) are L2-resident (192MB) -> matrix-pipe bound.
// Per-wave tile 32x32 (GEMM2) / 32x16 dual-matrix (GEMM1): 4 WMMA per
// 4 fragment loads; fragments loaded as 2x ds_load_b128 each.
// Staging uses GLOBAL_LOAD_ASYNC_TO_LDS_B128 (ASYNCcnt) when available.
// ---------------------------------------------------------------------------

typedef __attribute__((ext_vector_type(16))) __bf16 v16bf;
typedef __attribute__((ext_vector_type(8)))  float  v8f;

constexpr int Bc = 4, Sc = 1024, Hc = 1024, Ec = 8, Ic = 2048, TOPKc = 2;
constexpr int Tc = Bc * Sc;               // 4096 tokens
constexpr float SOFT_CAP = 30.0f;

#if defined(__has_builtin)
# if __has_builtin(__builtin_amdgcn_global_load_async_to_lds_b128)
#  define USE_ASYNC_LDS 1
# endif
#endif
#ifndef USE_ASYNC_LDS
# define USE_ASYNC_LDS 0
#endif

#if USE_ASYNC_LDS
#define AS1 __attribute__((address_space(1)))
#define AS3 __attribute__((address_space(3)))
typedef int v4i_ __attribute__((vector_size(16)));
__device__ __forceinline__ void async_cp16(const void* g, void* l) {
  __builtin_amdgcn_global_load_async_to_lds_b128((AS1 v4i_*)g,
                                                 (AS3 v4i_*)l, 0, 0);
}
__device__ __forceinline__ void async_wait0() {
#if defined(__has_builtin) && __has_builtin(__builtin_amdgcn_s_wait_asynccnt)
  __builtin_amdgcn_s_wait_asynccnt(0);
#else
  asm volatile("s_wait_asynccnt 0x0" ::: "memory");
#endif
}
#endif

__device__ __forceinline__ unsigned short f2bf(float f) {
  unsigned int u = __float_as_uint(f);
  u += 0x7fffu + ((u >> 16) & 1u);        // round-to-nearest-even
  return (unsigned short)(u >> 16);
}

// LDS rows are 36 dwords (32 data + 4 pad): 16B aligned rows, conflict-free
// fragment reads. A fragment's 8 dwords are two contiguous 4-dword runs.
constexpr int LDSW = 36;

union Frag { v16bf v; unsigned int u[8]; uint4 q[2]; };

__device__ __forceinline__ void load_frag(const unsigned int* __restrict__ row,
                                          int kt, int khalf, Frag& f) {
  const unsigned int* p = row + kt * 16 + khalf * 4;
  f.q[0] = *(const uint4*)(p);        // words {0..3}+khalf*4   -> ds_load_b128
  f.q[1] = *(const uint4*)(p + 8);    // words {8..11}+khalf*4  -> ds_load_b128
}

__device__ __forceinline__ v8f wmma_bf16(const Frag& a, const Frag& b, v8f c) {
  return __builtin_amdgcn_wmma_f32_16x16x32_bf16(false, a.v, false, b.v,
                                                 (short)0, c, false, false);
}

// ---------------- init: zero output + routing counters ---------------------
__global__ void init_kernel(float* __restrict__ out, int n,
                            int* __restrict__ counts, int* __restrict__ cursors) {
  if (blockIdx.x == 0 && threadIdx.x < Ec) {
    counts[threadIdx.x] = 0;
    cursors[threadIdx.x] = 0;
  }
  for (int i = blockIdx.x * blockDim.x + threadIdx.x; i < n;
       i += gridDim.x * blockDim.x)
    out[i] = 0.0f;
}

// ---------------- fp32 -> bf16 pack (vectorized) ---------------------------
__global__ void cvt4_kernel(const float4* __restrict__ src,
                            uint2* __restrict__ dst, int n4) {
  int i = blockIdx.x * blockDim.x + threadIdx.x;
  int st = gridDim.x * blockDim.x;
  for (; i < n4; i += st) {
    float4 v = src[i];
    uint2 o;
    o.x = (unsigned)f2bf(v.x) | ((unsigned)f2bf(v.y) << 16);
    o.y = (unsigned)f2bf(v.z) | ((unsigned)f2bf(v.w) << 16);
    dst[i] = o;
  }
}

// ---------------- router: logits -> softcap -> softmax -> top2 -------------
__global__ __launch_bounds__(128) void router_kernel(
    const float* __restrict__ x, const float* __restrict__ wg,
    int* __restrict__ counts, int* __restrict__ top_idx,
    float* __restrict__ top_w) {
  const int t = blockIdx.x;
  const int tid = threadIdx.x;           // 128 threads
  float p[Ec];
#pragma unroll
  for (int e = 0; e < Ec; ++e) p[e] = 0.0f;
  const float* xr = x + (size_t)t * Hc;
  for (int h = tid; h < Hc; h += 128) {
    float xv = xr[h];
#pragma unroll
    for (int e = 0; e < Ec; ++e) p[e] += xv * wg[e * Hc + h];
  }
  __shared__ float red[128][Ec];
#pragma unroll
  for (int e = 0; e < Ec; ++e) red[tid][e] = p[e];
  __syncthreads();
  __shared__ float logits[Ec];
  if (tid < Ec) {
    float s = 0.0f;
    for (int i = 0; i < 128; ++i) s += red[i][tid];
    logits[tid] = SOFT_CAP * tanhf(s / SOFT_CAP);
  }
  __syncthreads();
  if (tid == 0) {
    float mx = -1e30f;
#pragma unroll
    for (int e = 0; e < Ec; ++e) mx = fmaxf(mx, logits[e]);
    float pr[Ec];
    float denom = 0.0f;
#pragma unroll
    for (int e = 0; e < Ec; ++e) { pr[e] = __expf(logits[e] - mx); denom += pr[e]; }
    float inv = 1.0f / denom;
#pragma unroll
    for (int e = 0; e < Ec; ++e) pr[e] *= inv;
    int i0 = 0;
#pragma unroll
    for (int e = 1; e < Ec; ++e) if (pr[e] > pr[i0]) i0 = e;
    int i1 = (i0 == 0) ? 1 : 0;
#pragma unroll
    for (int e = 0; e < Ec; ++e) if (e != i0 && pr[e] > pr[i1]) i1 = e;
    top_idx[2 * t + 0] = i0; top_w[2 * t + 0] = pr[i0];
    top_idx[2 * t + 1] = i1; top_w[2 * t + 1] = pr[i1];
    atomicAdd(&counts[i0], 1);
    atomicAdd(&counts[i1], 1);
  }
}

// ---------------- exclusive prefix over 8 experts --------------------------
__global__ void scan_kernel(const int* __restrict__ counts, int* __restrict__ offsets) {
  if (blockIdx.x == 0 && threadIdx.x == 0) {
    int acc = 0;
    for (int e = 0; e < Ec; ++e) { offsets[e] = acc; acc += counts[e]; }
    offsets[Ec] = acc;                   // == T*TOPK
  }
}

// ---------------- compact (token, gate) rows per expert --------------------
__global__ void scatter_kernel(const int* __restrict__ top_idx,
                               const float* __restrict__ top_w,
                               const int* __restrict__ offsets,
                               int* __restrict__ cursors,
                               int* __restrict__ row_tok,
                               float* __restrict__ row_gate) {
  int t = blockIdx.x * blockDim.x + threadIdx.x;
  if (t >= Tc) return;
#pragma unroll
  for (int j = 0; j < TOPKc; ++j) {
    int e = top_idx[2 * t + j];
    int slot = atomicAdd(&cursors[e], 1);
    int row = offsets[e] + slot;
    row_tok[row] = t;
    row_gate[row] = top_w[2 * t + j];
  }
}

// ---------------- GEMM1: act = gelu(X W1^T) * (X W3^T)  (per expert) -------
// block: 256 thr = 8 waves; tile M=64 (mt 2x32), N=64 (nt 4x16); K-stage 64.
// wave: 2 A frags + 1 w1 frag + 1 w3 frag -> 4 WMMA per K-tile.
__global__ __launch_bounds__(256) void moe_gemm1_kernel(
    const unsigned short* __restrict__ xb,   // [T,H]   bf16
    const unsigned short* __restrict__ w1b,  // [E,I,H] bf16
    const unsigned short* __restrict__ w3b,  // [E,I,H] bf16
    const int* __restrict__ offsets,
    const int* __restrict__ row_tok,
    unsigned short* __restrict__ act) {      // [T*TOPK, I] bf16
  const int e = blockIdx.z;
  const int nblk = blockIdx.x;               // 64-wide tile in I
  const int mblk = blockIdx.y;               // 64-row tile
  const int base = offsets[e];
  const int rows = offsets[e + 1] - base;
  if (mblk * 64 >= rows) return;

  __shared__ __align__(16) unsigned int As[64][LDSW];
  __shared__ __align__(16) unsigned int W1s[64][LDSW];
  __shared__ __align__(16) unsigned int W3s[64][LDSW];

  const int tid = threadIdx.x;
  const int lane = tid & 31;
  const int wv = tid >> 5;
  const int mt = wv >> 2;                    // 0..1  (32 rows each)
  const int nt = wv & 3;                     // 0..3  (16 cols each)
  const int khalf = lane >> 4;
  const int lm = lane & 15;

  // staging: row = tid>>2 (0..63), 32B (16 bf16) per thread per K-stage
  const int srow = tid >> 2;
  const int scg = tid & 3;                   // col group: bf16 col = scg*16
  const int grow = mblk * 64 + srow;
  const int tok = row_tok[base + ((grow < rows) ? grow : (rows - 1))];
  const size_t abase = (size_t)tok * Hc + scg * 16;
  const size_t wbase = ((size_t)e * Ic + (size_t)(nblk * 64 + srow)) * Hc + scg * 16;

  v8f acc1[2] = {v8f{}, v8f{}};
  v8f acc3[2] = {v8f{}, v8f{}};

  for (int k0 = 0; k0 < Hc; k0 += 64) {
    if (k0 + 64 < Hc) {                      // gfx1250 global_prefetch_b8
      __builtin_prefetch(w1b + wbase + k0 + 64, 0, 3);
      __builtin_prefetch(w3b + wbase + k0 + 64, 0, 3);
    }
#if USE_ASYNC_LDS
    async_cp16(xb  + abase + k0,     &As [srow][scg * 8]);
    async_cp16(xb  + abase + k0 + 8, &As [srow][scg * 8 + 4]);
    async_cp16(w1b + wbase + k0,     &W1s[srow][scg * 8]);
    async_cp16(w1b + wbase + k0 + 8, &W1s[srow][scg * 8 + 4]);
    async_cp16(w3b + wbase + k0,     &W3s[srow][scg * 8]);
    async_cp16(w3b + wbase + k0 + 8, &W3s[srow][scg * 8 + 4]);
    async_wait0();                           // this wave's LDS writes landed
#else
    uint4 a0  = *(const uint4*)(xb  + abase + k0);
    uint4 a1  = *(const uint4*)(xb  + abase + k0 + 8);
    uint4 w10 = *(const uint4*)(w1b + wbase + k0);
    uint4 w11 = *(const uint4*)(w1b + wbase + k0 + 8);
    uint4 w30 = *(const uint4*)(w3b + wbase + k0);
    uint4 w31 = *(const uint4*)(w3b + wbase + k0 + 8);
    *(uint4*)&As [srow][scg * 8]     = a0;
    *(uint4*)&As [srow][scg * 8 + 4] = a1;
    *(uint4*)&W1s[srow][scg * 8]     = w10;
    *(uint4*)&W1s[srow][scg * 8 + 4] = w11;
    *(uint4*)&W3s[srow][scg * 8]     = w30;
    *(uint4*)&W3s[srow][scg * 8 + 4] = w31;
#endif
    __syncthreads();

#pragma unroll
    for (int kt = 0; kt < 2; ++kt) {
      Frag a0f, a1f, b1f, b3f;
      load_frag(&As [mt * 32 + lm][0],      kt, khalf, a0f);
      load_frag(&As [mt * 32 + 16 + lm][0], kt, khalf, a1f);
      load_frag(&W1s[nt * 16 + lm][0],      kt, khalf, b1f);
      load_frag(&W3s[nt * 16 + lm][0],      kt, khalf, b3f);
      acc1[0] = wmma_bf16(a0f, b1f, acc1[0]);
      acc1[1] = wmma_bf16(a1f, b1f, acc1[1]);
      acc3[0] = wmma_bf16(a0f, b3f, acc3[0]);
      acc3[1] = wmma_bf16(a1f, b3f, acc3[1]);
    }
    __syncthreads();
  }

  const int ncol = nblk * 64 + nt * 16 + lm;
#pragma unroll
  for (int half = 0; half < 2; ++half) {
#pragma unroll
    for (int v = 0; v < 8; ++v) {
      const int gr = mblk * 64 + mt * 32 + half * 16 + khalf * 8 + v;
      if (gr < rows) {
        float h1 = acc1[half][v], h3 = acc3[half][v];
        float g = 0.5f * h1 * (1.0f + erff(h1 * 0.70710678118f)); // exact GELU
        act[(size_t)(base + gr) * Ic + ncol] = f2bf(g * h3);
      }
    }
  }
}

// ---------------- GEMM2: out[t] += gate * (act W2^T)  (per expert) ---------
// block: 256 thr = 8 waves; tile M=64 (mt 2x32), N=128 (nt 4x32); K-stage 64.
// wave: 2 A frags + 2 B frags -> 4 WMMA per K-tile.
__global__ __launch_bounds__(256) void moe_gemm2_kernel(
    const unsigned short* __restrict__ act,  // [T*TOPK, I] bf16
    const unsigned short* __restrict__ w2b,  // [E,H,I]     bf16
    const int* __restrict__ offsets,
    const int* __restrict__ row_tok,
    const float* __restrict__ row_gate,
    float* __restrict__ out) {               // [T,H] fp32 (pre-zeroed)
  const int e = blockIdx.z;
  const int nblk = blockIdx.x;               // 128-wide tile in H
  const int mblk = blockIdx.y;               // 64-row tile
  const int base = offsets[e];
  const int rows = offsets[e + 1] - base;
  if (mblk * 64 >= rows) return;

  __shared__ __align__(16) unsigned int As[64][LDSW];
  __shared__ __align__(16) unsigned int Ws[128][LDSW];

  const int tid = threadIdx.x;
  const int lane = tid & 31;
  const int wv = tid >> 5;
  const int mt = wv >> 2;                    // 0..1 (32 rows)
  const int nt = wv & 3;                     // 0..3 (32 cols)
  const int khalf = lane >> 4;
  const int lm = lane & 15;

  // A staging: row = tid>>2 (0..63), 32B per thread per K-stage
  const int arow = tid >> 2;
  const int acg = tid & 3;
  const int grow = mblk * 64 + arow;
  const int arow_g = base + ((grow < rows) ? grow : (rows - 1));
  const size_t abase = (size_t)arow_g * Ic + acg * 16;
  // W staging: row = tid>>1 (0..127), 64B per thread per K-stage
  const int wrow = tid >> 1;
  const int wcg = tid & 1;                   // bf16 col = wcg*32
  const size_t wbase = ((size_t)e * Hc + (size_t)(nblk * 128 + wrow)) * Ic + wcg * 32;

  v8f acc[2][2] = {{v8f{}, v8f{}}, {v8f{}, v8f{}}};

  for (int k0 = 0; k0 < Ic; k0 += 64) {
    if (k0 + 64 < Ic) {                      // gfx1250 global_prefetch_b8
      __builtin_prefetch(w2b + wbase + k0 + 64, 0, 3);
    }
#if USE_ASYNC_LDS
    async_cp16(act + abase + k0,      &As[arow][acg * 8]);
    async_cp16(act + abase + k0 + 8,  &As[arow][acg * 8 + 4]);
    async_cp16(w2b + wbase + k0,      &Ws[wrow][wcg * 16]);
    async_cp16(w2b + wbase + k0 + 8,  &Ws[wrow][wcg * 16 + 4]);
    async_cp16(w2b + wbase + k0 + 16, &Ws[wrow][wcg * 16 + 8]);
    async_cp16(w2b + wbase + k0 + 24, &Ws[wrow][wcg * 16 + 12]);
    async_wait0();
#else
    uint4 a0 = *(const uint4*)(act + abase + k0);
    uint4 a1 = *(const uint4*)(act + abase + k0 + 8);
    uint4 b0 = *(const uint4*)(w2b + wbase + k0);
    uint4 b1 = *(const uint4*)(w2b + wbase + k0 + 8);
    uint4 b2 = *(const uint4*)(w2b + wbase + k0 + 16);
    uint4 b3 = *(const uint4*)(w2b + wbase + k0 + 24);
    *(uint4*)&As[arow][acg * 8]      = a0;
    *(uint4*)&As[arow][acg * 8 + 4]  = a1;
    *(uint4*)&Ws[wrow][wcg * 16]     = b0;
    *(uint4*)&Ws[wrow][wcg * 16 + 4] = b1;
    *(uint4*)&Ws[wrow][wcg * 16 + 8] = b2;
    *(uint4*)&Ws[wrow][wcg * 16 + 12]= b3;
#endif
    __syncthreads();

#pragma unroll
    for (int kt = 0; kt < 2; ++kt) {
      Frag a0f, a1f, b0f, b1f;
      load_frag(&As[mt * 32 + lm][0],           kt, khalf, a0f);
      load_frag(&As[mt * 32 + 16 + lm][0],      kt, khalf, a1f);
      load_frag(&Ws[nt * 32 + lm][0],           kt, khalf, b0f);
      load_frag(&Ws[nt * 32 + 16 + lm][0],      kt, khalf, b1f);
      acc[0][0] = wmma_bf16(a0f, b0f, acc[0][0]);
      acc[0][1] = wmma_bf16(a0f, b1f, acc[0][1]);
      acc[1][0] = wmma_bf16(a1f, b0f, acc[1][0]);
      acc[1][1] = wmma_bf16(a1f, b1f, acc[1][1]);
    }
    __syncthreads();
  }

#pragma unroll
  for (int mh = 0; mh < 2; ++mh) {
#pragma unroll
    for (int v = 0; v < 8; ++v) {
      const int gr = mblk * 64 + mt * 32 + mh * 16 + khalf * 8 + v;
      if (gr < rows) {
        const int t = row_tok[base + gr];
        const float g = row_gate[base + gr];
#pragma unroll
        for (int nh = 0; nh < 2; ++nh) {
          const int ncol = nblk * 128 + nt * 32 + nh * 16 + lm;
          atomicAdd(&out[(size_t)t * Hc + ncol], g * acc[mh][nh][v]);
        }
      }
    }
  }
}

// ---------------------------------------------------------------------------
extern "C" void kernel_launch(void* const* d_in, const int* in_sizes, int n_in,
                              void* d_out, int out_size, void* d_ws, size_t ws_size,
                              hipStream_t stream) {
  (void)in_sizes; (void)n_in; (void)out_size; (void)ws_size;
  const float* x  = (const float*)d_in[0];   // [B,S,H]
  const float* wg = (const float*)d_in[1];   // [E,H]
  const float* w1 = (const float*)d_in[2];   // [E,I,H]
  const float* w3 = (const float*)d_in[3];   // [E,I,H]
  const float* w2 = (const float*)d_in[4];   // [E,H,I]
  // d_in[5] = top_k (constant 2, baked in)
  float* out = (float*)d_out;

  char* ws = (char*)d_ws;
  size_t off = 0;
  auto alloc = [&](size_t bytes) -> char* {
    char* p = ws + off;
    off = (off + bytes + 255) & ~(size_t)255;
    return p;
  };
  unsigned short* xb  = (unsigned short*)alloc((size_t)Tc * Hc * 2);        //  8 MB
  unsigned short* w1b = (unsigned short*)alloc((size_t)Ec * Ic * Hc * 2);   // 32 MB
  unsigned short* w3b = (unsigned short*)alloc((size_t)Ec * Ic * Hc * 2);   // 32 MB
  unsigned short* w2b = (unsigned short*)alloc((size_t)Ec * Hc * Ic * 2);   // 32 MB
  unsigned short* act = (unsigned short*)alloc((size_t)Tc * TOPKc * Ic * 2);// 32 MB
  int*   counts  = (int*)alloc(256);
  int*   offsets = (int*)alloc(256);
  int*   cursors = (int*)alloc(256);
  int*   top_idx = (int*)alloc((size_t)Tc * TOPKc * 4);
  float* top_w   = (float*)alloc((size_t)Tc * TOPKc * 4);
  int*   row_tok = (int*)alloc((size_t)Tc * TOPKc * 4);
  float* row_gate= (float*)alloc((size_t)Tc * TOPKc * 4);

  init_kernel<<<2048, 256, 0, stream>>>(out, Tc * Hc, counts, cursors);
  cvt4_kernel<<<4096, 256, 0, stream>>>((const float4*)x,  (uint2*)xb,  Tc * Hc / 4);
  cvt4_kernel<<<4096, 256, 0, stream>>>((const float4*)w1, (uint2*)w1b, Ec * Ic * Hc / 4);
  cvt4_kernel<<<4096, 256, 0, stream>>>((const float4*)w3, (uint2*)w3b, Ec * Ic * Hc / 4);
  cvt4_kernel<<<4096, 256, 0, stream>>>((const float4*)w2, (uint2*)w2b, Ec * Hc * Ic / 4);
  router_kernel<<<Tc, 128, 0, stream>>>(x, wg, counts, top_idx, top_w);
  scan_kernel<<<1, 32, 0, stream>>>(counts, offsets);
  scatter_kernel<<<(Tc + 255) / 256, 256, 0, stream>>>(top_idx, top_w, offsets,
                                                       cursors, row_tok, row_gate);
  dim3 g1(Ic / 64, (Tc + 63) / 64, Ec);
  moe_gemm1_kernel<<<g1, 256, 0, stream>>>(xb, w1b, w3b, offsets, row_tok, act);
  dim3 g2(Hc / 128, (Tc + 63) / 64, Ec);
  moe_gemm2_kernel<<<g2, 256, 0, stream>>>(act, w2b, offsets, row_tok, row_gate, out);
}
